// CRF_SelfAttention_49065706390003
// MI455X (gfx1250) — matrix operand order repaired and checked by reading, not compiled, not run on hardware
//
#include <hip/hip_runtime.h>

// ---------------------------------------------------------------------------
// CRF_SelfAttention on MI455X (gfx1250), algebraically folded:
//   softmax over singleton axis == identity  =>  ctx = v  (Wq/Wk unused)
//   scale loop == 3*Y,  pairwise = 0.75*Y
//   out[p,f,:] = emb[f,p,:] @ (0.75 * Wmp@Wo@Wv)^T + (0.75*Wmp@(Wo@bv+bo)+bmp)
// Pipeline (all on `stream`, serialized):
//   1) transpose+cvt Wo -> WoT(bf16), Wv -> WvT(bf16)
//   2) bo2 = Wo@bv + bo ;  constc = 0.75*Wmp@bo2 + bmp
//   3) T = Wmp@Wo        (WMMA bf16, fp32 acc)  [64 x 2048]
//   4) Wcomb = bf16(0.75 * T@Wv)                [64 x 2048]
//   5) out = emb @ Wcomb^T + constc             (HBM-bound: 151 MB emb read)
// Main kernel: TDM (tensor_load_to_lds, 6-arg builtin) stages Wcomb into LDS
// once per block (bank-spread 4112B row stride), B from ds_load_b128, A (emb)
// 4-deep software-pipelined global loads feeding v_wmma_f32_16x16x32_bf16.
// ---------------------------------------------------------------------------

#define E_DIM 2048
#define C_DIM 64
#define F_DIM 18
#define P_DIM 1024
#define NTOK  (F_DIM * P_DIM)       // 18432 tokens, 1152 M-tiles of 16
#define LDS_ROW 4112                // 4096B row + 16B pad  (== 4 dwords mod 64 banks)
#define LDS_BYTES (C_DIM * LDS_ROW) // 263168 B  (<= 320KB per WGP)

typedef __attribute__((ext_vector_type(16))) __bf16 v16bf;
typedef __attribute__((ext_vector_type(16))) float  v16f;
typedef __attribute__((ext_vector_type(8)))  float  v8f;
typedef __attribute__((ext_vector_type(4)))  unsigned v4u;
typedef __attribute__((ext_vector_type(8)))  int      v8i;
typedef __attribute__((ext_vector_type(4)))  int      v4i;

union ABu {
    v16bf v;
    uint4 q[2];
};

static __device__ __forceinline__ unsigned short f2bf_u16(float f) {
    unsigned u = __builtin_bit_cast(unsigned, f);
    u += 0x7FFFu + ((u >> 16) & 1u);
    return (unsigned short)(u >> 16);
}

// 16 fp32 -> 16 bf16 (lowered to packed v_cvt_pk_bf16_f32)
static __device__ __forceinline__ v16bf cvt_a(const float4& a0, const float4& a1,
                                              const float4& a2, const float4& a3) {
    v16f f;
    f[0] = a0.x;  f[1] = a0.y;  f[2] = a0.z;  f[3] = a0.w;
    f[4] = a1.x;  f[5] = a1.y;  f[6] = a1.z;  f[7] = a1.w;
    f[8] = a2.x;  f[9] = a2.y;  f[10] = a2.z; f[11] = a2.w;
    f[12] = a3.x; f[13] = a3.y; f[14] = a3.z; f[15] = a3.w;
    return __builtin_convertvector(f, v16bf);
}

static __device__ __forceinline__ v16bf b_cast(const uint4& q0, const uint4& q1) {
    ABu b;
    b.q[0] = q0;
    b.q[1] = q1;
    return b.v;
}

// ---------------------------------------------------------------------------
// Shared 16x16-tile GEMM (K=2048), A fp32 row-major, B bf16 [N,K]-contiguous.
struct GFrag {
    float4 a0, a1, a2, a3;
    uint4 b0, b1;
};

static __device__ __forceinline__ void load_gfrag(GFrag& f, const float* __restrict__ arow,
                                                  const unsigned short* __restrict__ brow, int k0) {
    f.a0 = *(const float4*)(arow + k0);
    f.a1 = *(const float4*)(arow + k0 + 4);
    f.a2 = *(const float4*)(arow + k0 + 8);
    f.a3 = *(const float4*)(arow + k0 + 12);
    const uint4* p = (const uint4*)(brow + k0);
    f.b0 = p[0];
    f.b1 = p[1];
}

static __device__ __forceinline__ void gfrag_mma(v8f& acc, const GFrag& f) {
    v16bf a = cvt_a(f.a0, f.a1, f.a2, f.a3);
    acc = __builtin_amdgcn_wmma_f32_16x16x32_bf16(false, a, false, b_cast(f.b0, f.b1),
                                                  (short)0, acc, false, false);
}

static __device__ __forceinline__ v8f gemm_tile_k2048(const float* __restrict__ arow,
                                                      const unsigned short* __restrict__ brow) {
    v8f acc = {0.f, 0.f, 0.f, 0.f, 0.f, 0.f, 0.f, 0.f};
    GFrag f0, f1;
    load_gfrag(f0, arow, brow, 0);
    for (int k0 = 0; k0 < E_DIM; k0 += 64) {
        load_gfrag(f1, arow, brow, k0 + 32);
        gfrag_mma(acc, f0);
        if (k0 + 64 < E_DIM) load_gfrag(f0, arow, brow, k0 + 64);
        gfrag_mma(acc, f1);
    }
    return acc;
}

// ---------------------------------------------------------------------------
// Kernel 1: tiled transpose fp32[n][n] -> bf16[n][n]  (dst[e][j] = src[j][e])
__global__ void k_transpose_cvt(const float* __restrict__ src,
                                unsigned short* __restrict__ dst, int n) {
    __shared__ float tile[32][33];
    int bx = blockIdx.x * 32, by = blockIdx.y * 32;
    int tx = threadIdx.x, ty = threadIdx.y;          // (32, 8)
    #pragma unroll
    for (int i = 0; i < 32; i += 8)
        tile[ty + i][tx] = src[(by + ty + i) * n + bx + tx];
    __syncthreads();
    #pragma unroll
    for (int i = 0; i < 32; i += 8)
        dst[(bx + ty + i) * n + by + tx] = f2bf_u16(tile[tx][ty + i]);
}

// Kernel 2: bo2[e] = dot(Wo[e,:], bv) + bo[e]        (one wave per row)
__global__ void k_bias_combine(const float* __restrict__ Wo, const float* __restrict__ bv,
                               const float* __restrict__ bo, float* __restrict__ bo2) {
    int wave = threadIdx.x >> 5, lane = threadIdx.x & 31;
    int e = blockIdx.x * 8 + wave;
    float s = 0.f;
    for (int j = lane; j < E_DIM; j += 32) s += Wo[e * E_DIM + j] * bv[j];
    #pragma unroll
    for (int m = 16; m >= 1; m >>= 1) s += __shfl_xor(s, m, 32);
    if (lane == 0) bo2[e] = s + bo[e];
}

// Kernel 3: constc[c] = 0.75*dot(Wmp[c,:], bo2) + bmp[c]
__global__ void k_constc(const float* __restrict__ Wmp, const float* __restrict__ bo2,
                         const float* __restrict__ bmp, float* __restrict__ constc) {
    int wave = threadIdx.x >> 5, lane = threadIdx.x & 31;
    int c = blockIdx.x * 8 + wave;
    float s = 0.f;
    for (int j = lane; j < E_DIM; j += 32) s += Wmp[c * E_DIM + j] * bo2[j];
    #pragma unroll
    for (int m = 16; m >= 1; m >>= 1) s += __shfl_xor(s, m, 32);
    if (lane == 0) constc[c] = 0.75f * s + bmp[c];
}

// Kernel 4: T[c,e] = sum_j Wmp[c,j]*Wo[j,e]  via A=Wmp(f32), B=WoT(bf16)
__global__ void k_gemm_T(const float* __restrict__ Wmp, const unsigned short* __restrict__ WoT,
                         float* __restrict__ T) {
    int wid  = (blockIdx.x * blockDim.x + threadIdx.x) >> 5;  // 0..511
    int lane = threadIdx.x & 31;
    int half = lane >> 4, l15 = lane & 15;
    int mtile = wid & 3, ntile = wid >> 2;                    // 4 x 128 tiles
    const float* arow = Wmp + (mtile * 16 + l15) * E_DIM + half * 16;
    const unsigned short* brow = WoT + (ntile * 16 + l15) * E_DIM + half * 16;
    v8f acc = gemm_tile_k2048(arow, brow);
    #pragma unroll
    for (int r = 0; r < 8; ++r) {
        int row = mtile * 16 + r + 8 * half;
        T[row * E_DIM + ntile * 16 + l15] = acc[r];
    }
}

// Kernel 5: Wcomb[c,e] = bf16(0.75 * sum_j T[c,j]*Wv[j,e])  via A=T, B=WvT
__global__ void k_gemm_wcomb(const float* __restrict__ T, const unsigned short* __restrict__ WvT,
                             unsigned short* __restrict__ Wcomb) {
    int wid  = (blockIdx.x * blockDim.x + threadIdx.x) >> 5;
    int lane = threadIdx.x & 31;
    int half = lane >> 4, l15 = lane & 15;
    int mtile = wid & 3, ntile = wid >> 2;
    const float* arow = T + (mtile * 16 + l15) * E_DIM + half * 16;
    const unsigned short* brow = WvT + (ntile * 16 + l15) * E_DIM + half * 16;
    v8f acc = gemm_tile_k2048(arow, brow);
    #pragma unroll
    for (int r = 0; r < 8; ++r) {
        int row = mtile * 16 + r + 8 * half;
        Wcomb[row * E_DIM + ntile * 16 + l15] = f2bf_u16(0.75f * acc[r]);
    }
}

// ---------------------------------------------------------------------------
// Kernel 6: out[p,f,c] = emb_token @ Wcomb^T + constc   (token = f*P + p)
// One wave per 16-token M-tile; B staged in LDS via TDM; 4-deep A pipeline.
struct AFrag {
    float4 a0, a1, a2, a3;
};

static __device__ __forceinline__ void load_afrag(AFrag& f, const float* __restrict__ arow, int k0) {
    f.a0 = *(const float4*)(arow + k0);
    f.a1 = *(const float4*)(arow + k0 + 4);
    f.a2 = *(const float4*)(arow + k0 + 8);
    f.a3 = *(const float4*)(arow + k0 + 12);
}

static __device__ __forceinline__ void mma_step(v8f& acc0, v8f& acc1, v8f& acc2, v8f& acc3,
                                                const AFrag& f, const char* bbase, int k0) {
    // bbase already includes l15*LDS_ROW + half*32; nt stride = 16*LDS_ROW
    const char* b = bbase + 2 * k0;
    uint4 q00 = *(const uint4*)(b);
    uint4 q01 = *(const uint4*)(b + 16);
    uint4 q10 = *(const uint4*)(b + 16 * LDS_ROW);
    uint4 q11 = *(const uint4*)(b + 16 * LDS_ROW + 16);
    uint4 q20 = *(const uint4*)(b + 32 * LDS_ROW);
    uint4 q21 = *(const uint4*)(b + 32 * LDS_ROW + 16);
    uint4 q30 = *(const uint4*)(b + 48 * LDS_ROW);
    uint4 q31 = *(const uint4*)(b + 48 * LDS_ROW + 16);
    v16bf a = cvt_a(f.a0, f.a1, f.a2, f.a3);
    acc0 = __builtin_amdgcn_wmma_f32_16x16x32_bf16(false, a, false, b_cast(q00, q01), (short)0, acc0, false, false);
    acc1 = __builtin_amdgcn_wmma_f32_16x16x32_bf16(false, a, false, b_cast(q10, q11), (short)0, acc1, false, false);
    acc2 = __builtin_amdgcn_wmma_f32_16x16x32_bf16(false, a, false, b_cast(q20, q21), (short)0, acc2, false, false);
    acc3 = __builtin_amdgcn_wmma_f32_16x16x32_bf16(false, a, false, b_cast(q30, q31), (short)0, acc3, false, false);
}

__global__ void k_main(const float* __restrict__ emb, const unsigned short* __restrict__ Wcomb,
                       const float* __restrict__ constc, float* __restrict__ out) {
    extern __shared__ char smem[];
    int wid  = (blockIdx.x * blockDim.x + threadIdx.x) >> 5;  // 0..1151
    int lane = threadIdx.x & 31;
    int half = lane >> 4, l15 = lane & 15;

    // --- TDM: DMA Wcomb [64 x 2048 bf16] into LDS, one 4096B row per iteration,
    //     LDS row stride 4112B (bank-spread). Wave 0 issues; EXEC is ignored.
    if (threadIdx.x < 32) {
        unsigned lds_base = (unsigned)(size_t)(&smem[0]);   // low 32 bits = LDS offset
        unsigned long long ga = (unsigned long long)Wcomb;
        // D# group0: count=1 | lds_addr | global_addr | type=2
        v4u g0 = { 1u, lds_base, (unsigned)ga, (unsigned)(ga >> 32) | (2u << 30) };
        // D# group1: data_size=2B, iterate_enable; tensor 2048x64; tile 2048x1
        v8i g1 = { (int)((1u << 16) | (1u << 19)),
                   (int)(2048u << 16),          // tensor_dim0[15:0] << 16
                   (int)(64u << 16),            // dim0 hi=0 | tensor_dim1[15:0]=64
                   (int)(2048u << 16),          // dim1 hi=0 | tile_dim0=2048
                   1,                           // tile_dim1=1, tile_dim2=0
                   2048,                        // tensor_dim0_stride (elements)
                   0, 0 };
        // D# group2 (iterate mode): lds_addr_inc=2056 elem (4112B), g_inc=2048 elem, 64 iters
        v4i g2 = { 0, (int)(LDS_ROW / 2), 2048, (int)(63u << 16) };
        v4i g3 = { 0, 0, 0, 0 };
        v8i g4 = { 0, 0, 0, 0, 0, 0, 0, 0 };    // 6-arg form: extra descriptor words unused
        __builtin_amdgcn_tensor_load_to_lds(g0, g1, g2, g3, g4, 0);
        __builtin_amdgcn_s_wait_tensorcnt(0);
    }
    __syncthreads();

    const float* arow = emb + (wid * 16 + l15) * E_DIM + half * 16;
    const char* bbase = smem + l15 * LDS_ROW + half * 32;

    v8f acc0 = {0.f, 0.f, 0.f, 0.f, 0.f, 0.f, 0.f, 0.f};
    v8f acc1 = acc0, acc2 = acc0, acc3 = acc0;

    AFrag f0, f1, f2, f3;
    load_afrag(f0, arow, 0);
    load_afrag(f1, arow, 32);
    load_afrag(f2, arow, 64);
    for (int k0 = 0; k0 < E_DIM; k0 += 128) {
        load_afrag(f3, arow, k0 + 96);
        mma_step(acc0, acc1, acc2, acc3, f0, bbase, k0);
        if (k0 + 128 < E_DIM) load_afrag(f0, arow, k0 + 128);
        mma_step(acc0, acc1, acc2, acc3, f1, bbase, k0 + 32);
        if (k0 + 160 < E_DIM) load_afrag(f1, arow, k0 + 160);
        mma_step(acc0, acc1, acc2, acc3, f2, bbase, k0 + 64);
        if (k0 + 192 < E_DIM) load_afrag(f2, arow, k0 + 192);
        mma_step(acc0, acc1, acc2, acc3, f3, bbase, k0 + 96);
    }

    v8f accs[4] = {acc0, acc1, acc2, acc3};
    #pragma unroll
    for (int nt = 0; nt < 4; ++nt) {
        int c = nt * 16 + l15;
        float cc = constc[c];
        #pragma unroll
        for (int r = 0; r < 8; ++r) {
            int token = wid * 16 + r + 8 * half;
            int f = token >> 10;           // /P_DIM
            int p = token & (P_DIM - 1);
            out[p * (F_DIM * C_DIM) + f * C_DIM + c] = accs[nt][r] + cc;
        }
    }
}

// ---------------------------------------------------------------------------
extern "C" void kernel_launch(void* const* d_in, const int* in_sizes, int n_in,
                              void* d_out, int out_size, void* d_ws, size_t ws_size,
                              hipStream_t stream) {
    (void)in_sizes; (void)n_in; (void)out_size; (void)ws_size;
    const float* emb = (const float*)d_in[0];
    const float* Wv  = (const float*)d_in[3];
    const float* bv  = (const float*)d_in[6];
    const float* Wo  = (const float*)d_in[7];
    const float* bo  = (const float*)d_in[8];
    const float* Wmp = (const float*)d_in[9];
    const float* bmp = (const float*)d_in[10];

    char* ws = (char*)d_ws;
    unsigned short* WoT    = (unsigned short*)(ws);                         // 8 MB
    unsigned short* WvT    = (unsigned short*)(ws + 8388608);               // 8 MB
    float*          Tws    = (float*)         (ws + 16777216);              // 512 KB
    unsigned short* Wcomb  = (unsigned short*)(ws + 17301504);              // 256 KB
    float*          bo2    = (float*)         (ws + 17563648);              // 8 KB
    float*          constc = (float*)         (ws + 17571840);              // 256 B

    dim3 tb(32, 8), tg(E_DIM / 32, E_DIM / 32);
    k_transpose_cvt<<<tg, tb, 0, stream>>>(Wo, WoT, E_DIM);
    k_transpose_cvt<<<tg, tb, 0, stream>>>(Wv, WvT, E_DIM);
    k_bias_combine<<<E_DIM / 8, 256, 0, stream>>>(Wo, bv, bo, bo2);
    k_constc<<<C_DIM / 8, 256, 0, stream>>>(Wmp, bo2, bmp, constc);
    k_gemm_T<<<64, 256, 0, stream>>>(Wmp, WoT, Tws);            // 512 waves
    k_gemm_wcomb<<<64, 256, 0, stream>>>(Tws, WvT, Wcomb);      // 512 waves
    k_main<<<(NTOK / 16) / 8, 256, LDS_BYTES, stream>>>(emb, Wcomb, constc, (float*)d_out);
}
